// SelectorConditionedLocalEvidenceEncoderV3413_14869176779187
// MI455X (gfx1250) — matrix operand order repaired and checked by reading, not compiled, not run on hardware
//
#include <hip/hip_runtime.h>
#include <hip/hip_bf16.h>

typedef __attribute__((ext_vector_type(16))) _Float16 v16h;
typedef __attribute__((ext_vector_type(8)))  _Float16 v8h;
typedef __attribute__((ext_vector_type(8)))  float    v8f;

#define B_ 8
#define M_ 128
#define T_ 32
#define H_ 64
#define U_ 32
#define D_ 256
#define S_ 384
#define QIN 516
#define QPAD 544
#define UIN 262
#define UPAD 288

// output offsets (floats) in concatenated return-order tuple
#define O_LEV   ((size_t)0)
#define O_LSEM  ((size_t)16777216)
#define O_ATTN  ((size_t)41943040)
#define O_USAGE ((size_t)44040192)
#define O_UEV   ((size_t)44105728)
#define O_UUS   ((size_t)48300032)
#define O_ENT   ((size_t)48316416)
#define O_MAXA  ((size_t)48381952)
#define O_SAL   ((size_t)48447488)

__device__ __forceinline__ float fclean(float x){
  return (__builtin_isnan(x) || __builtin_isinf(x)) ? 0.0f : x;
}
__device__ __forceinline__ float gelu_exact(float x){
  return 0.5f * x * (1.0f + erff(x * 0.70710678118654752f));
}
__device__ __forceinline__ float clip01(float x){ return fminf(fmaxf(x, 0.f), 1.f); }

// A-fragment: 16-bit A 16x32 — lanes 0-15 hold K=0..15, lanes 16-31 hold K=16..31 (contiguous)
__device__ __forceinline__ v16h ldfragA(const _Float16* p){
  v8h lo = *(const v8h*)p;
  v8h hi = *(const v8h*)(p + 8);
  v16h r;
#pragma unroll
  for (int i = 0; i < 8; ++i){ r[i] = lo[i]; r[8 + i] = hi[i]; }
  return r;
}
// B-fragment: 16-bit B 32x16 — lanes 0-15 hold K=0..15, lanes 16-31 hold K=16..31 (contiguous)
__device__ __forceinline__ v16h ldfragB(const _Float16* p){
  v8h lo = *(const v8h*)p;
  v8h hi = *(const v8h*)(p + 8);
  v16h r;
#pragma unroll
  for (int i = 0; i < 8; ++i){ r[i] = lo[i]; r[8 + i] = hi[i]; }
  return r;
}
__device__ __forceinline__ v8f wmma32f16(v16h a, v16h b, v8f c){
  return __builtin_amdgcn_wmma_f32_16x16x32_f16(false, a, false, b, (short)0, c, false, false);
}
__device__ __forceinline__ float rsum8(float v){
#pragma unroll
  for (int m = 1; m < 8; m <<= 1) v += __shfl_xor(v, m, 32);
  return v;
}
__device__ __forceinline__ float rsum16(float v){
#pragma unroll
  for (int m = 1; m < 16; m <<= 1) v += __shfl_xor(v, m, 32);
  return v;
}
__device__ __forceinline__ float rmax16(float v){
#pragma unroll
  for (int m = 1; m < 16; m <<= 1) v = fmaxf(v, __shfl_xor(v, m, 32));
  return v;
}

// ---------------- K0: weight convert fp32[K][N] -> f16 transposed [N][Kpad] ----------------
__global__ void k_wconv(const float* __restrict__ src, _Float16* __restrict__ dst,
                        int Kin, int Kpad, int Nout){
  int idx = blockIdx.x * blockDim.x + threadIdx.x;
  int tot = Nout * Kpad;
  if (idx >= tot) return;
  int n = idx / Kpad, k = idx % Kpad;
  dst[idx] = (k < Kin) ? (_Float16)src[(size_t)k * Nout + n] : (_Float16)0.0f;
}

// ---------------- K1: key/value MLPs (LN -> 384x256 gelu -> 256x256) ----------------
__global__ void __launch_bounds__(128)
k_kv_mlp(const float* __restrict__ sem,
         const float* __restrict__ lnk_g, const float* __restrict__ lnk_b,
         const float* __restrict__ lnv_g, const float* __restrict__ lnv_b,
         const _Float16* __restrict__ wk1t, const _Float16* __restrict__ wk2t,
         const _Float16* __restrict__ wv1t, const _Float16* __restrict__ wv2t,
         const float* __restrict__ bk1, const float* __restrict__ bk2,
         const float* __restrict__ bv1, const float* __restrict__ bv2,
         _Float16* __restrict__ keyF, _Float16* __restrict__ valF)
{
  __shared__ _Float16 xh[2][16][S_];
  __shared__ _Float16 hid[2][16][D_];
  const int tid = threadIdx.x, lane = tid & 31, wave = tid >> 5;
  const int r0 = blockIdx.x * 16;
  {
    int row = tid >> 3, g = tid & 7;
    const float* x = sem + (size_t)(r0 + row) * S_;
    float s = 0.f, ss = 0.f;
    for (int k = g; k < S_; k += 8){ float v = fclean(x[k]); s += v; ss += v * v; }
    float se = rsum8(s), sse = rsum8(ss);
    float mu = se / S_;
    float rstd = rsqrtf(fmaxf(sse / S_ - mu * mu, 0.f) + 1e-5f);
    for (int k = g; k < S_; k += 8){
      float v = (fclean(x[k]) - mu) * rstd;
      xh[0][row][k] = (_Float16)(v * lnk_g[k] + lnk_b[k]);
      xh[1][row][k] = (_Float16)(v * lnv_g[k] + lnv_b[k]);
    }
  }
  __syncthreads();
  const int kbA = (lane >> 4) * 16, kbB = (lane >> 4) * 16, rb = (lane >> 4) * 8;
  for (int jj = wave; jj < 32; jj += 4){
    int br = jj >> 4, ct = jj & 15;
    const _Float16* W = br ? wv1t : wk1t;
    const float*    b1 = br ? bv1 : bk1;
    int col = ct * 16 + (lane & 15);
    v8f acc = (v8f)(b1[col]);
    const _Float16* arow = &xh[br][lane & 15][kbA];
    const _Float16* brow = W + (size_t)col * S_ + kbB;
#pragma unroll
    for (int kc = 0; kc < 12; ++kc)
      acc = wmma32f16(ldfragA(arow + kc * 32), ldfragB(brow + kc * 32), acc);
#pragma unroll
    for (int j = 0; j < 8; ++j)
      hid[br][rb + j][col] = (_Float16)gelu_exact(acc[j]);
  }
  __syncthreads();
  for (int jj = wave; jj < 32; jj += 4){
    int br = jj >> 4, ct = jj & 15;
    const _Float16* W = br ? wv2t : wk2t;
    const float*    b2 = br ? bv2 : bk2;
    int col = ct * 16 + (lane & 15);
    v8f acc = (v8f)(b2[col]);
    const _Float16* arow = &hid[br][lane & 15][kbA];
    const _Float16* brow = W + (size_t)col * D_ + kbB;
#pragma unroll
    for (int kc = 0; kc < 8; ++kc)
      acc = wmma32f16(ldfragA(arow + kc * 32), ldfragB(brow + kc * 32), acc);
    _Float16* o = br ? valF : keyF;
#pragma unroll
    for (int j = 0; j < 8; ++j)
      o[(size_t)(r0 + rb + j) * D_ + col] = (_Float16)acc[j];
  }
}

// ---------------- K2: query MLP (build q_in, LN over 516 -> 544x256 gelu -> 256x256) ----------------
__device__ __forceinline__ float qin_elem(int k, const float* pt, const float* sh,
                                          float hpos, const float* sc){
  if (k < 256)  return pt[k];
  if (k < 512)  return sh[k - 256];
  if (k == 512) return hpos;
  if (k == 513) return sc[0];
  if (k == 514) return sc[1];
  return sc[2];
}

__global__ void __launch_bounds__(128)
k_q_mlp(const float* __restrict__ pt, const float* __restrict__ sh,
        const float* __restrict__ conf, const float* __restrict__ agree,
        const float* __restrict__ vis, const float* __restrict__ selc,
        const float* __restrict__ lnq_g, const float* __restrict__ lnq_b,
        const _Float16* __restrict__ wq1t, const _Float16* __restrict__ wq2t,
        const float* __restrict__ bq1, const float* __restrict__ bq2,
        _Float16* __restrict__ qryF)
{
  __shared__ _Float16 qA[16][QPAD];
  __shared__ _Float16 hid[16][D_];
  __shared__ float sc[3];
  const int tid = threadIdx.x, lane = tid & 31, wave = tid >> 5;
  const int r0 = blockIdx.x * 16;
  const int bm = r0 / H_;
  const int b = bm / M_;
  if (tid == 0){
    float cs = 0.f, as = 0.f, vs = 0.f;
    const float* cp = conf + (size_t)bm * T_;
    const float* ap = agree + (size_t)bm * T_;
    const float* vp = vis + (size_t)bm * T_;
    for (int t = 0; t < T_; ++t){ cs += clip01(cp[t]); as += clip01(ap[t]); vs += clip01(vp[t]); }
    sc[0] = cs / T_;
    sc[1] = (as / T_) * (vs / T_);
    sc[2] = clip01(selc[bm]);
  }
  __syncthreads();
  {
    int row = tid >> 3, g = tid & 7;
    int h = (r0 + row) % H_;
    const float* ptr = pt + (size_t)bm * D_;
    const float* shr = sh + (size_t)(b * H_ + h) * D_;
    float hpos = (float)h / (H_ - 1);
    float s = 0.f, ss = 0.f;
    for (int k = g; k < QIN; k += 8){
      float v = qin_elem(k, ptr, shr, hpos, sc);
      s += v; ss += v * v;
    }
    float se = rsum8(s), sse = rsum8(ss);
    float mu = se / QIN;
    float rstd = rsqrtf(fmaxf(sse / QIN - mu * mu, 0.f) + 1e-5f);
    for (int k = g; k < QIN; k += 8){
      float v = qin_elem(k, ptr, shr, hpos, sc);
      qA[row][k] = (_Float16)((v - mu) * rstd * lnq_g[k] + lnq_b[k]);
    }
    for (int k = QIN + g; k < QPAD; k += 8) qA[row][k] = (_Float16)0.0f;
  }
  __syncthreads();
  const int kbA = (lane >> 4) * 16, kbB = (lane >> 4) * 16, rb = (lane >> 4) * 8;
  for (int ct = wave; ct < 16; ct += 4){
    int col = ct * 16 + (lane & 15);
    v8f acc = (v8f)(bq1[col]);
    const _Float16* arow = &qA[lane & 15][kbA];
    const _Float16* brow = wq1t + (size_t)col * QPAD + kbB;
#pragma unroll
    for (int kc = 0; kc < 17; ++kc)
      acc = wmma32f16(ldfragA(arow + kc * 32), ldfragB(brow + kc * 32), acc);
#pragma unroll
    for (int j = 0; j < 8; ++j)
      hid[rb + j][col] = (_Float16)gelu_exact(acc[j]);
  }
  __syncthreads();
  for (int ct = wave; ct < 16; ct += 4){
    int col = ct * 16 + (lane & 15);
    v8f acc = (v8f)(bq2[col]);
    const _Float16* arow = &hid[lane & 15][kbA];
    const _Float16* brow = wq2t + (size_t)col * D_ + kbB;
#pragma unroll
    for (int kc = 0; kc < 8; ++kc)
      acc = wmma32f16(ldfragA(arow + kc * 32), ldfragB(brow + kc * 32), acc);
#pragma unroll
    for (int j = 0; j < 8; ++j)
      qryF[(size_t)(r0 + rb + j) * D_ + col] = (_Float16)acc[j];
  }
}

// ---------------- K3: attention per (b,m): QK^T, softmax+stats, attn@V ----------------
__global__ void __launch_bounds__(128)
k_attn(const _Float16* __restrict__ keyF, const _Float16* __restrict__ valF,
       const _Float16* __restrict__ qryF,
       const int* __restrict__ maskp, const float* __restrict__ conf,
       const float* __restrict__ agree, const float* __restrict__ selw_raw,
       float* __restrict__ out)
{
  __shared__ _Float16 keyL[T_][D_];
  __shared__ _Float16 valT[D_][T_];
  __shared__ _Float16 attnH[H_][T_];
  __shared__ float lbias[T_], lmask[T_], lsw[T_];
  const int tid = threadIdx.x, lane = tid & 31, wave = tid >> 5;
  const int bm = blockIdx.x;
  const size_t kvbase = (size_t)bm * T_ * D_;
  for (int i = tid; i < (T_ * D_) / 8; i += 128)
    *(v8h*)(&keyL[0][0] + i * 8) = *(const v8h*)(keyF + kvbase + (size_t)i * 8);
  for (int i = tid; i < T_ * D_; i += 128){
    int t = i / D_, d = i % D_;
    valT[d][t] = valF[kvbase + i];
  }
  if (tid < T_){
    int t = tid;
    float mf = maskp[(size_t)bm * T_ + t] ? 1.0f : 0.0f;
    float sw = fmaxf(selw_raw[(size_t)bm * T_ + t], 0.0f) * mf;
    float ssum = sw;
#pragma unroll
    for (int m2 = 1; m2 < 32; m2 <<= 1) ssum += __shfl_xor(ssum, m2, 32);
    float swn = sw / fmaxf(ssum, 1e-6f);
    lsw[t] = swn;
    lmask[t] = mf;
    lbias[t] = clip01(conf[(size_t)bm * T_ + t]) + clip01(agree[(size_t)bm * T_ + t])
             + 0.15f * ((float)t / (T_ - 1)) + 2.0f * logf(fmaxf(swn, 1e-6f));
  }
  __syncthreads();
  const int kbA = (lane >> 4) * 16, kbB = (lane >> 4) * 16, rb = (lane >> 4) * 8;
  const size_t qbase = (size_t)bm * H_ * D_;
  v8f l0 = (v8f)0.0f, l1 = (v8f)0.0f;
  {
    const _Float16* arow = qryF + qbase + (size_t)(wave * 16 + (lane & 15)) * D_ + kbA;
    const _Float16* bp0 = &keyL[lane & 15][kbB];
    const _Float16* bp1 = &keyL[16 + (lane & 15)][kbB];
#pragma unroll
    for (int kc = 0; kc < 8; ++kc){
      v16h a = ldfragA(arow + kc * 32);
      l0 = wmma32f16(a, ldfragB(bp0 + kc * 32), l0);
      l1 = wmma32f16(a, ldfragB(bp1 + kc * 32), l1);
    }
  }
  const int t0 = lane & 15, t1 = 16 + (lane & 15);
  const float bias0 = lbias[t0], bias1 = lbias[t1];
  const float m0 = lmask[t0], m1 = lmask[t1];
  const float sw0 = lsw[t0], sw1 = lsw[t1];
  float attn0[8], attn1[8];
#pragma unroll
  for (int j = 0; j < 8; ++j){
    float x0 = (m0 > 0.f) ? (l0[j] * 0.0625f + bias0) : -10000.0f;
    float x1 = (m1 > 0.f) ? (l1[j] * 0.0625f + bias1) : -10000.0f;
    float mx = rmax16(fmaxf(x0, x1));
    float p0 = expf(x0 - mx) * m0;
    float p1 = expf(x1 - mx) * m1;
    float sm = rsum16(p0 + p1);
    float rs = 1.0f / fmaxf(sm, 1e-6f);
    attn0[j] = p0 * rs; attn1[j] = p1 * rs;
  }
  const size_t oh = (size_t)bm * H_;
#pragma unroll
  for (int j = 0; j < 8; ++j){
    int h = wave * 16 + rb + j;
    float a0 = attn0[j], a1 = attn1[j];
    out[O_ATTN + (oh + h) * T_ + t0] = a0;
    out[O_ATTN + (oh + h) * T_ + t1] = a1;
    attnH[h][t0] = (_Float16)a0;
    attnH[h][t1] = (_Float16)a1;
    float ac0 = fmaxf(a0, 1e-8f), ac1 = fmaxf(a1, 1e-8f);
    float ent = rsum16(ac0 * logf(ac0) + ac1 * logf(ac1));
    float mxa = rmax16(fmaxf(a0, a1));
    float sal = rsum16(a0 * sw0 + a1 * sw1);
    if ((lane & 15) == 0){
      out[O_ENT  + oh + h] = -ent * (1.0f / logf(32.0f));
      out[O_MAXA + oh + h] = mxa;
      out[O_SAL  + oh + h] = sal;
    }
  }
  __syncthreads();
  {
    v16h a = ldfragA(&attnH[wave * 16 + (lane & 15)][kbA]);
#pragma unroll
    for (int ct = 0; ct < 16; ++ct){
      int col = ct * 16 + (lane & 15);
      v8f acc = (v8f)0.0f;
      acc = wmma32f16(a, ldfragB(&valT[col][kbB]), acc);
#pragma unroll
      for (int j = 0; j < 8; ++j){
        int h = wave * 16 + rb + j;
        out[O_LEV + (oh + h) * D_ + col] = acc[j];
      }
    }
  }
}

// ---------------- K4: sem projection + normalize, usage head ----------------
__global__ void __launch_bounds__(128)
k_sem_usage(const float* __restrict__ lne_g, const float* __restrict__ lne_b,
            const float* __restrict__ lnu_g, const float* __restrict__ lnu_b,
            const _Float16* __restrict__ wet, const _Float16* __restrict__ wu1t,
            const float* __restrict__ be, const float* __restrict__ bu1,
            const float* __restrict__ wu2, const float* __restrict__ bu2,
            const float* __restrict__ conf, const float* __restrict__ agree,
            const float* __restrict__ selc,
            float* __restrict__ out)
{
  __shared__ _Float16 Ae[16][D_];
  __shared__ _Float16 Au[16][UPAD];
  __shared__ float semO[16][S_];
  __shared__ _Float16 uh[16][D_];
  __shared__ float sc[3];
  const int tid = threadIdx.x, lane = tid & 31, wave = tid >> 5;
  const int r0 = blockIdx.x * 16;
  const int bm = r0 / H_;
  if (tid == 0){
    float cs = 0.f, as = 0.f;
    for (int t = 0; t < T_; ++t){ cs += clip01(conf[(size_t)bm * T_ + t]); as += clip01(agree[(size_t)bm * T_ + t]); }
    sc[0] = cs / T_; sc[1] = as / T_; sc[2] = clip01(selc[bm]);
  }
  __syncthreads();
  {
    int row = tid >> 3, g = tid & 7;
    size_t r = (size_t)r0 + row;
    const float* lev = out + O_LEV + r * D_;
    float s = 0.f, ss = 0.f;
    for (int k = g; k < D_; k += 8){ float v = lev[k]; s += v; ss += v * v; }
    float se = rsum8(s), sse = rsum8(ss);
    float mu = se / D_;
    float rstd = rsqrtf(fmaxf(sse / D_ - mu * mu, 0.f) + 1e-5f);
    for (int k = g; k < D_; k += 8)
      Ae[row][k] = (_Float16)((lev[k] - mu) * rstd * lne_g[k] + lne_b[k]);
    float ext[6];
    ext[0] = out[O_ENT + r]; ext[1] = out[O_MAXA + r]; ext[2] = out[O_SAL + r];
    ext[3] = sc[0]; ext[4] = sc[1]; ext[5] = sc[2];
    float s2 = s, ss2 = ss;
    for (int k = 256 + g; k < UIN; k += 8){ float v = ext[k - 256]; s2 += v; ss2 += v * v; }
    float su = rsum8(s2), ssu = rsum8(ss2);
    float mu2 = su / UIN;
    float rstd2 = rsqrtf(fmaxf(ssu / UIN - mu2 * mu2, 0.f) + 1e-5f);
    for (int k = g; k < D_; k += 8)
      Au[row][k] = (_Float16)((lev[k] - mu2) * rstd2 * lnu_g[k] + lnu_b[k]);
    for (int k = 256 + g; k < UIN; k += 8)
      Au[row][k] = (_Float16)((ext[k - 256] - mu2) * rstd2 * lnu_g[k] + lnu_b[k]);
    for (int k = UIN + g; k < UPAD; k += 8) Au[row][k] = (_Float16)0.0f;
  }
  __syncthreads();
  const int kbA = (lane >> 4) * 16, kbB = (lane >> 4) * 16, rb = (lane >> 4) * 8;
  for (int ct = wave; ct < 24; ct += 4){
    int col = ct * 16 + (lane & 15);
    v8f acc = (v8f)(be[col]);
    const _Float16* arow = &Ae[lane & 15][kbA];
    const _Float16* brow = wet + (size_t)col * D_ + kbB;
#pragma unroll
    for (int kc = 0; kc < 8; ++kc)
      acc = wmma32f16(ldfragA(arow + kc * 32), ldfragB(brow + kc * 32), acc);
#pragma unroll
    for (int j = 0; j < 8; ++j) semO[rb + j][col] = acc[j];
  }
  for (int ct = wave; ct < 16; ct += 4){
    int col = ct * 16 + (lane & 15);
    v8f acc = (v8f)(bu1[col]);
    const _Float16* arow = &Au[lane & 15][kbA];
    const _Float16* brow = wu1t + (size_t)col * UPAD + kbB;
#pragma unroll
    for (int kc = 0; kc < 9; ++kc)
      acc = wmma32f16(ldfragA(arow + kc * 32), ldfragB(brow + kc * 32), acc);
#pragma unroll
    for (int j = 0; j < 8; ++j) uh[rb + j][col] = (_Float16)gelu_exact(acc[j]);
  }
  __syncthreads();
  {
    int row = tid >> 3, g = tid & 7;
    size_t r = (size_t)r0 + row;
    float ss = 0.f;
    for (int k = g; k < S_; k += 8){ float v = semO[row][k]; ss += v * v; }
    ss = rsum8(ss);
    float rn = 1.0f / fmaxf(sqrtf(ss), 1e-12f);
    float* dst = out + O_LSEM + r * S_;
    for (int k = g; k < S_; k += 8) dst[k] = semO[row][k] * rn;
    float dot = 0.f;
    for (int k = g; k < D_; k += 8) dot += (float)uh[row][k] * wu2[k];
    dot = rsum8(dot);
    if (g == 0){
      float z = dot + bu2[0];
      out[O_USAGE + r] = 1.0f / (1.0f + expf(-z));
    }
  }
}

// ---------------- K5: unit_evidence = assign^T @ local_evidence / denom ----------------
__global__ void __launch_bounds__(256)
k_unit_ev(const float* __restrict__ assign, const float* __restrict__ lev,
          float* __restrict__ uev)
{
  __shared__ float aw[16][U_];
  __shared__ float rden[U_];
  const int tid = threadIdx.x;
  const int b = blockIdx.x / H_, h = blockIdx.x % H_;
  if (tid < U_){
    float s = 0.f;
    for (int m = 0; m < M_; ++m) s += assign[(size_t)(b * M_ + m) * U_ + tid];
    rden[tid] = 1.0f / fmaxf(s, 1e-6f);
  }
  float acc[U_];
#pragma unroll
  for (int u = 0; u < U_; ++u) acc[u] = 0.f;
  const int d = tid;
  for (int mc = 0; mc < M_; mc += 16){
    __syncthreads();
    for (int i = tid; i < 16 * U_; i += 256){
      int mi = i / U_, u = i % U_;
      aw[mi][u] = assign[(size_t)(b * M_ + mc + mi) * U_ + u];
    }
    __syncthreads();
    for (int mi = 0; mi < 16; ++mi){
      float lv = lev[((size_t)(b * M_ + mc + mi) * H_ + h) * D_ + d];
#pragma unroll
      for (int u = 0; u < U_; ++u) acc[u] += aw[mi][u] * lv;
    }
  }
  __syncthreads();
#pragma unroll
  for (int u = 0; u < U_; ++u)
    uev[((size_t)(b * U_ + u) * H_ + h) * D_ + d] = acc[u] * rden[u];
}

// ---------------- K6: unit_usage ----------------
__global__ void k_unit_usage(const float* __restrict__ assign, const float* __restrict__ usage,
                             float* __restrict__ uus)
{
  int idx = blockIdx.x * blockDim.x + threadIdx.x;
  if (idx >= B_ * U_ * H_) return;
  int b = idx / (U_ * H_);
  int u = (idx / H_) % U_;
  int h = idx % H_;
  float s = 0.f, den = 0.f;
  for (int m = 0; m < M_; ++m){
    float a = assign[(size_t)(b * M_ + m) * U_ + u];
    den += a;
    s += a * usage[(size_t)(b * M_ + m) * H_ + h];
  }
  uus[idx] = clip01(s / fmaxf(den, 1e-6f));
}

extern "C" void kernel_launch(void* const* d_in, const int* in_sizes, int n_in,
                              void* d_out, int out_size, void* d_ws, size_t ws_size,
                              hipStream_t stream)
{
  (void)in_sizes; (void)n_in; (void)out_size; (void)ws_size;
  // params (dict order) then tensors
  const float* lnv_g = (const float*)d_in[0];
  const float* lnv_b = (const float*)d_in[1];
  const float* wv1   = (const float*)d_in[2];
  const float* bv1   = (const float*)d_in[3];
  const float* wv2   = (const float*)d_in[4];
  const float* bv2   = (const float*)d_in[5];
  const float* lnk_g = (const float*)d_in[6];
  const float* lnk_b = (const float*)d_in[7];
  const float* wk1   = (const float*)d_in[8];
  const float* bk1   = (const float*)d_in[9];
  const float* wk2   = (const float*)d_in[10];
  const float* bk2   = (const float*)d_in[11];
  const float* lnq_g = (const float*)d_in[12];
  const float* lnq_b = (const float*)d_in[13];
  const float* wq1   = (const float*)d_in[14];
  const float* bq1   = (const float*)d_in[15];
  const float* wq2   = (const float*)d_in[16];
  const float* bq2   = (const float*)d_in[17];
  const float* lnu_g = (const float*)d_in[18];
  const float* lnu_b = (const float*)d_in[19];
  const float* wu1   = (const float*)d_in[20];
  const float* bu1   = (const float*)d_in[21];
  const float* wu2   = (const float*)d_in[22];
  const float* bu2   = (const float*)d_in[23];
  const float* lne_g = (const float*)d_in[24];
  const float* lne_b = (const float*)d_in[25];
  const float* we    = (const float*)d_in[26];
  const float* be    = (const float*)d_in[27];
  const float* pt    = (const float*)d_in[28];
  const float* sh    = (const float*)d_in[29];
  const float* sem   = (const float*)d_in[30];
  const int*   maskp = (const int*)d_in[31];
  const float* conf  = (const float*)d_in[32];
  const float* agree = (const float*)d_in[33];
  const float* vis   = (const float*)d_in[34];
  const float* assign= (const float*)d_in[35];
  const float* selw  = (const float*)d_in[36];
  const float* selc  = (const float*)d_in[37];
  float* out = (float*)d_out;

  char* w = (char*)d_ws;
  auto take = [&](size_t bytes) -> void* {
    void* p = (void*)w;
    w += (bytes + 255) & ~(size_t)255;
    return p;
  };
  _Float16* wk1t = (_Float16*)take((size_t)256 * 384 * 2);
  _Float16* wk2t = (_Float16*)take((size_t)256 * 256 * 2);
  _Float16* wv1t = (_Float16*)take((size_t)256 * 384 * 2);
  _Float16* wv2t = (_Float16*)take((size_t)256 * 256 * 2);
  _Float16* wq1t = (_Float16*)take((size_t)256 * QPAD * 2);
  _Float16* wq2t = (_Float16*)take((size_t)256 * 256 * 2);
  _Float16* wu1t = (_Float16*)take((size_t)256 * UPAD * 2);
  _Float16* wet  = (_Float16*)take((size_t)384 * 256 * 2);
  _Float16* keyF = (_Float16*)take((size_t)B_ * M_ * T_ * D_ * 2);
  _Float16* valF = (_Float16*)take((size_t)B_ * M_ * T_ * D_ * 2);
  _Float16* qryF = (_Float16*)take((size_t)B_ * M_ * H_ * D_ * 2);

  auto conv = [&](const float* src, _Float16* dst, int Kin, int Kpad, int Nout){
    int tot = Nout * Kpad;
    k_wconv<<<(tot + 255) / 256, 256, 0, stream>>>(src, dst, Kin, Kpad, Nout);
  };
  conv(wk1, wk1t, 384, 384, 256);
  conv(wk2, wk2t, 256, 256, 256);
  conv(wv1, wv1t, 384, 384, 256);
  conv(wv2, wv2t, 256, 256, 256);
  conv(wq1, wq1t, QIN, QPAD, 256);
  conv(wq2, wq2t, 256, 256, 256);
  conv(wu1, wu1t, UIN, UPAD, 256);
  conv(we,  wet,  256, 256, 384);

  k_kv_mlp<<<(B_ * M_ * T_) / 16, 128, 0, stream>>>(
      sem, lnk_g, lnk_b, lnv_g, lnv_b, wk1t, wk2t, wv1t, wv2t,
      bk1, bk2, bv1, bv2, keyF, valF);

  k_q_mlp<<<(B_ * M_ * H_) / 16, 128, 0, stream>>>(
      pt, sh, conf, agree, vis, selc, lnq_g, lnq_b, wq1t, wq2t, bq1, bq2, qryF);

  k_attn<<<B_ * M_, 128, 0, stream>>>(
      keyF, valF, qryF, maskp, conf, agree, selw, out);

  k_sem_usage<<<(B_ * M_ * H_) / 16, 128, 0, stream>>>(
      lne_g, lne_b, lnu_g, lnu_b, wet, wu1t, be, bu1, wu2, bu2,
      conf, agree, selc, out);

  k_unit_ev<<<B_ * H_, 256, 0, stream>>>(assign, out + O_LEV, out + O_UEV);

  k_unit_usage<<<(B_ * U_ * H_ + 255) / 256, 256, 0, stream>>>(
      assign, out + O_USAGE, out + O_UUS);
}
